// CellGNN_56951266345692
// MI455X (gfx1250) — matrix operand order repaired and checked by reading, not compiled, not run on hardware
//
#include <hip/hip_runtime.h>

#define N_NODES 50000
#define N_EDGES 800000
#define EMB 2
#define HID 128
#define IN_SZ 9
#define INV_MAX_RADIUS 10.0f

typedef __attribute__((ext_vector_type(16))) _Float16 v16h;
typedef __attribute__((ext_vector_type(8)))  float    v8f;

union Frag16 {
    v16h v;
    uint4 q[2];
    _Float16 h[16];
};

// Packed-weight buffer offsets (in f16 halves). Fragment = 32 lanes * 16 halves = 512 halves (1KB).
#define OFF_L0 0                    // 8 frags  (kt=0, nt=0..7), K padded 9->32
#define OFF_L1 (OFF_L0 + 8  * 512)  // 32 frags (kt*8+nt): kt0..1 then kt2..3
#define OFF_L2 (OFF_L1 + 32 * 512)
#define OFF_L3 (OFF_L2 + 32 * 512)
#define OFF_L4 (OFF_L3 + 32 * 512)  // 4 frags  (kt=0..3, nt=0), N padded 2->16
#define TOTAL_HALVES (OFF_L4 + 4 * 512)

#define HALF_CHUNK (16 * 512)       // 16 fragments = 16KB = one pipeline chunk

// ---------------------------------------------------------------------------
__global__ void zero_kernel(float* __restrict__ p, int n) {
    int i = blockIdx.x * 256 + threadIdx.x;
    if (i < n) p[i] = 0.0f;
}

// ---------------------------------------------------------------------------
// f32 weights -> f16, pre-swizzled into WMMA B-matrix lane layout:
//   lane 0..15 : col n = lane,    K = 0..7,16..23 ; lane 16..31 : col n = lane-16, K = 8..15,24..31
// ---------------------------------------------------------------------------
__global__ void pack_weights(const float* __restrict__ w0, const float* __restrict__ w1,
                             const float* __restrict__ w2, const float* __restrict__ w3,
                             const float* __restrict__ w4, _Float16* __restrict__ wp) {
    int idx = blockIdx.x * 256 + threadIdx.x;
    if (idx >= TOTAL_HALVES) return;
    int frag   = idx >> 9;
    int within = idx & 511;
    int lane   = within >> 4;
    int h      = within & 15;
    int n16    = lane & 15;
    int khalf  = ((h & 8) << 1) + ((lane >> 4) << 3) + (h & 7);    // K within 32-deep block

    float val = 0.0f;
    if (frag < 8) {
        int k = khalf, n = frag * 16 + n16;
        if (k < IN_SZ) val = w0[k * HID + n];
    } else if (frag < 40) {
        int f = frag - 8, kt = f >> 3, nt = f & 7;
        val = w1[(kt * 32 + khalf) * HID + nt * 16 + n16];
    } else if (frag < 72) {
        int f = frag - 40, kt = f >> 3, nt = f & 7;
        val = w2[(kt * 32 + khalf) * HID + nt * 16 + n16];
    } else if (frag < 104) {
        int f = frag - 72, kt = f >> 3, nt = f & 7;
        val = w3[(kt * 32 + khalf) * HID + nt * 16 + n16];
    } else {
        int kt = frag - 104;
        if (n16 < 2) val = w4[(kt * 32 + khalf) * 2 + n16];
    }
    wp[idx] = (_Float16)val;
}

// ---------------------------------------------------------------------------
// CDNA5 async global->LDS staging (ASYNCcnt-tracked, bypasses VGPRs).
// One instruction per wave moves 32 lanes x 16B = 512B. Trip count is a
// compile-time constant so issues are straight-line with full EXEC and the
// per-wave ASYNCcnt contribution is statically known (HALVES/8/256).
// ---------------------------------------------------------------------------
template <int HALVES>
__device__ __forceinline__ void issue_chunk(const _Float16* __restrict__ gsrc,
                                            _Float16* sdst, int tid) {
    constexpr int NITER = (HALVES / 8) / 256;      // async instrs per wave
    static_assert(NITER * 256 * 8 == HALVES, "chunk must tile evenly");
    const uint4* g = (const uint4*)gsrc + tid;
    unsigned lds = (unsigned)(size_t)(sdst + tid * 8);
#pragma unroll
    for (int j = 0; j < NITER; ++j) {
        asm volatile("global_load_async_to_lds_b128 %0, %1, off"
                     :: "v"(lds), "v"(g) : "memory");
        g += 256;
        lds += 256 * 16;
    }
}

template <int N>
__device__ __forceinline__ void wait_async() {
    asm volatile("s_wait_asynccnt %0" :: "i"(N) : "memory");
}

// ---------------------------------------------------------------------------
// WMMA fragment loaders (documented CDNA5 16-bit layouts).
// ---------------------------------------------------------------------------
__device__ __forceinline__ Frag16 loadA(const _Float16 (*act)[HID], int mstripe, int kt, int lane) {
    Frag16 A;
    int row = mstripe * 16 + (lane & 15);
    int c0  = kt * 32 + ((lane >> 4) << 3);      // +0 (lanes 0-15) or +8 (lanes 16-31)
    const uint4* p = (const uint4*)&act[row][c0];
    A.q[0] = p[0];                               // halves K = c0    .. c0+7
    A.q[1] = p[2];                               // halves K = c0+16 .. c0+23
    return A;
}

__device__ __forceinline__ Frag16 loadB(const _Float16* w, int lane) {
    Frag16 B;
    const uint4* p = (const uint4*)(w + lane * 16);
    B.q[0] = p[0];
    B.q[1] = p[1];
    return B;
}

__device__ __forceinline__ v8f wmma16(const Frag16& A, const Frag16& B, v8f c) {
    return __builtin_amdgcn_wmma_f32_16x16x32_f16(false, A.v, false, B.v, (short)0, c,
                                                  false, false);
}

// half-layer: 2 k-steps x 4 n-tiles from a 16-fragment buffer
__device__ __forceinline__ void mma_half(const _Float16 (*act)[HID], const _Float16* wbuf,
                                         int kbase, int mstripe, int ntbase, int lane,
                                         v8f* acc) {
#pragma unroll
    for (int kk = 0; kk < 2; ++kk) {
        Frag16 A = loadA(act, mstripe, kbase + kk, lane);
#pragma unroll
        for (int t = 0; t < 4; ++t) {
            Frag16 B = loadB(wbuf + (kk * 8 + ntbase + t) * 512, lane);
            acc[t] = wmma16(A, B, acc[t]);
        }
    }
}

// bias + ReLU + f16 writeback into activation buffer
__device__ __forceinline__ void store_act(_Float16 (*act)[HID], const v8f* acc,
                                          const float* __restrict__ bias,
                                          int mstripe, int ntbase, int lane) {
    int rbase = mstripe * 16 + ((lane >> 4) << 3);
    int c0    = lane & 15;
#pragma unroll
    for (int t = 0; t < 4; ++t) {
        int col = (ntbase + t) * 16 + c0;
        float b = bias[col];
#pragma unroll
        for (int v = 0; v < 8; ++v) {
            float x = fmaxf(acc[t][v] + b, 0.0f);
            act[rbase + v][col] = (_Float16)x;
        }
    }
}

// ---------------------------------------------------------------------------
// Main kernel: 64 edges/WG, 8 waves. Wave pair owns one 16-row M-stripe;
// each wave does 4 of 8 N-tiles. Weight chunks C0..C7 stream asynchronously
// into double-buffered LDS while the previous chunk feeds the WMMAs.
//   per-wave issues: C0=L0(2)  C1=L1a(4)  C2=L1b(4)  C3=L2a(4)
//                    C4=L2b(4) C5=L3a(4)  C6=L3b(4)  C7=L4(1)
// ---------------------------------------------------------------------------
__global__ __launch_bounds__(256) void boids_mlp_kernel(
    const float* __restrict__ pos, const float* __restrict__ vel,
    const float* __restrict__ aemb, const float* __restrict__ vnorm,
    const int* __restrict__ data_id, const long long* __restrict__ eidx,
    const _Float16* __restrict__ wp,
    const float* __restrict__ b0, const float* __restrict__ b1,
    const float* __restrict__ b2, const float* __restrict__ b3,
    const float* __restrict__ b4, float* __restrict__ out)
{
    __shared__ __align__(16) _Float16 sAct[64][HID];      // 16 KB activations
    __shared__ __align__(16) _Float16 sW[2][HALF_CHUNK];  // 2 x 16 KB weight buffers
    __shared__ int sDst[64];

    const int tid     = threadIdx.x;
    const int wave    = tid >> 5;
    const int lane    = tid & 31;
    const int mstripe = wave >> 1;
    const int ntbase  = (wave & 1) * 4;
    const int e0      = blockIdx.x * 64;

    // ---- prologue: start streaming L0 and L1a while features are built ----
    issue_chunk<8 * 512>(wp + OFF_L0, &sW[0][0], tid);        // C0 -> buf0
    issue_chunk<HALF_CHUNK>(wp + OFF_L1, &sW[1][0], tid);     // C1 -> buf1

    if (tid < 64) {
        int e = e0 + tid;
        float f[IN_SZ];
#pragma unroll
        for (int k = 0; k < IN_SZ; ++k) f[k] = 0.0f;
        int dsti = -1;
        if (e < N_EDGES) {
            int d = (int)eidx[e];
            int s = (int)eidx[N_EDGES + e];
            dsti = d;
            float dx = (pos[2 * s]     - pos[2 * d])     * INV_MAX_RADIUS;
            float dy = (pos[2 * s + 1] - pos[2 * d + 1]) * INV_MAX_RADIUS;
            float inv = 1.0f / vnorm[0];
            int did = data_id[0];
            f[0] = dx;
            f[1] = dy;
            f[2] = sqrtf(dx * dx + dy * dy);
            f[3] = vel[2 * d] * inv;
            f[4] = vel[2 * d + 1] * inv;
            f[5] = vel[2 * s] * inv;
            f[6] = vel[2 * s + 1] * inv;
            const float* ap = aemb + ((long long)did * N_NODES + d) * EMB;
            f[7] = ap[0];
            f[8] = ap[1];
        }
        sDst[tid] = dsti;
#pragma unroll
        for (int k = 0; k < IN_SZ; ++k) sAct[tid][k] = (_Float16)f[k];
#pragma unroll
        for (int k = IN_SZ; k < 32; ++k) sAct[tid][k] = (_Float16)0.0f;
        // warm L2 for the next tile's edge indices
        if (e + 64 < N_EDGES) {
            __builtin_prefetch(&eidx[e + 64], 0, 0);
            __builtin_prefetch(&eidx[N_EDGES + e + 64], 0, 0);
        }
    }

    v8f z = {};
    v8f acc[4];

    // ---- Layer 0 (buf0, single k-step) ----
    wait_async<4>();                  // C0 done (C1 still in flight)
    __syncthreads();                  // all waves' slices landed + features visible
#pragma unroll
    for (int t = 0; t < 4; ++t) acc[t] = z;
    {
        Frag16 A = loadA(sAct, mstripe, 0, lane);
#pragma unroll
        for (int t = 0; t < 4; ++t) {
            Frag16 B = loadB(&sW[0][0] + (ntbase + t) * 512, lane);
            acc[t] = wmma16(A, B, acc[t]);
        }
    }
    __syncthreads();                  // buf0 + sAct reads complete
    issue_chunk<HALF_CHUNK>(wp + OFF_L1 + HALF_CHUNK, &sW[0][0], tid);  // C2 -> buf0
    store_act(sAct, acc, b0, mstripe, ntbase, lane);
    __syncthreads();                  // activations visible

    // ---- Layers 1 and 2 (identical pipeline shape) ----
    for (int l = 0; l < 2; ++l) {
#pragma unroll
        for (int t = 0; t < 4; ++t) acc[t] = z;
        wait_async<4>();              // chunk in buf1 (C1 / C3) done
        __syncthreads();
        mma_half(sAct, &sW[1][0], 0, mstripe, ntbase, lane, acc);
        __syncthreads();              // buf1 reads done
        issue_chunk<HALF_CHUNK>((l == 0 ? wp + OFF_L2 : wp + OFF_L3), &sW[1][0], tid);
        wait_async<4>();              // chunk in buf0 (C2 / C4) done
        __syncthreads();
        mma_half(sAct, &sW[0][0], 2, mstripe, ntbase, lane, acc);
        __syncthreads();              // buf0 + sAct reads done
        issue_chunk<HALF_CHUNK>((l == 0 ? wp + OFF_L2 + HALF_CHUNK : wp + OFF_L3 + HALF_CHUNK),
                                &sW[0][0], tid);
        store_act(sAct, acc, (l == 0 ? b1 : b2), mstripe, ntbase, lane);
        __syncthreads();
    }

    // ---- Layer 3 (last async chunk is the small L4 block) ----
#pragma unroll
    for (int t = 0; t < 4; ++t) acc[t] = z;
    wait_async<4>();                  // C5 done
    __syncthreads();
    mma_half(sAct, &sW[1][0], 0, mstripe, ntbase, lane, acc);
    __syncthreads();
    issue_chunk<4 * 512>(wp + OFF_L4, &sW[1][0], tid);   // C7 -> buf1
    wait_async<1>();                  // C6 done (only C7 outstanding)
    __syncthreads();
    mma_half(sAct, &sW[0][0], 2, mstripe, ntbase, lane, acc);
    __syncthreads();
    store_act(sAct, acc, b3, mstripe, ntbase, lane);
    __syncthreads();

    // ---- Layer 4: N padded 2->16; even waves compute, then scatter-add ----
    wait_async<0>();                  // C7 done
    __syncthreads();
    if ((wave & 1) == 0) {
        v8f accf = {};
#pragma unroll
        for (int kt = 0; kt < 4; ++kt) {
            Frag16 A = loadA(sAct, mstripe, kt, lane);
            Frag16 B = loadB(&sW[1][0] + kt * 512, lane);
            accf = wmma16(A, B, accf);
        }
        int n = lane & 15;
        if (n < 2) {
            float bb = b4[n];
            int rbase = mstripe * 16 + ((lane >> 4) << 3);
#pragma unroll
            for (int v = 0; v < 8; ++v) {
                int dst = sDst[rbase + v];
                if (dst >= 0) atomicAdd(out + 2 * dst + n, accf[v] + bb);
            }
        }
    }
}

// ---------------------------------------------------------------------------
extern "C" void kernel_launch(void* const* d_in, const int* in_sizes, int n_in,
                              void* d_out, int out_size, void* d_ws, size_t ws_size,
                              hipStream_t stream) {
    (void)in_sizes; (void)n_in; (void)ws_size;
    const float*     pos     = (const float*)d_in[0];
    const float*     vel     = (const float*)d_in[1];
    const float*     aemb    = (const float*)d_in[2];
    const float*     vnorm   = (const float*)d_in[3];
    const float*     w0      = (const float*)d_in[4];
    const float*     b0      = (const float*)d_in[5];
    const float*     w1      = (const float*)d_in[6];
    const float*     b1      = (const float*)d_in[7];
    const float*     w2      = (const float*)d_in[8];
    const float*     b2      = (const float*)d_in[9];
    const float*     w3      = (const float*)d_in[10];
    const float*     b3      = (const float*)d_in[11];
    const float*     w4      = (const float*)d_in[12];
    const float*     b4      = (const float*)d_in[13];
    const long long* eidx    = (const long long*)d_in[14];
    const int*       data_id = (const int*)d_in[15];
    float*           out     = (float*)d_out;
    _Float16*        wp      = (_Float16*)d_ws;   // 110592 B packed f16 weights

    zero_kernel<<<(out_size + 255) / 256, 256, 0, stream>>>(out, out_size);
    pack_weights<<<(TOTAL_HALVES + 255) / 256, 256, 0, stream>>>(w0, w1, w2, w3, w4, wp);
    boids_mlp_kernel<<<N_EDGES / 64, 256, 0, stream>>>(
        pos, vel, aemb, vnorm, data_id, eidx, wp, b0, b1, b2, b3, b4, out);
}
